// QKVAttentionBlock_31482110279926
// MI455X (gfx1250) — matrix operand order repaired and checked by reading, not compiled, not run on hardware
//
#include <hip/hip_runtime.h>
#include <hip/hip_bf16.h>
#include <cstdint>

typedef _Float16 half_t;
typedef _Float16 v16h __attribute__((ext_vector_type(16)));
typedef _Float16 v8h  __attribute__((ext_vector_type(8)));
typedef _Float16 v4h  __attribute__((ext_vector_type(4)));
typedef float    v8f  __attribute__((ext_vector_type(8)));
typedef float    v4f  __attribute__((ext_vector_type(4)));
typedef unsigned int v4u __attribute__((ext_vector_type(4)));
typedef int          v4i __attribute__((ext_vector_type(4)));
typedef int          v8i __attribute__((ext_vector_type(8)));

#if defined(__has_builtin)
#if __has_builtin(__builtin_amdgcn_tensor_load_to_lds) && \
    __has_builtin(__builtin_amdgcn_s_wait_tensorcnt)
#define HAVE_TDM 1
#endif
#endif
#ifndef HAVE_TDM
#define HAVE_TDM 0
#endif

#define SEQ     4096
#define DIM     1024
#define KSPLIT  4
#define BM      64          // query rows per workgroup
#define BN      64          // kv rows per tile
#define NTHR    512         // 16 wave32
#define KEYS_PER_SPLIT (SEQ / KSPLIT)
#define NTILES  (KEYS_PER_SPLIT / BN)

// LDS strides (halves). Q/K rows: 1024 data + 4-half pad at col 512 + 4 at end
// (matches TDM pad_interval=1024B, pad_amount=2 dwords). V^T/P rows: 64 + 8.
#define LDQ  (DIM + 8)
#define LDK  (DIM + 8)
#define LDV  (BN + 8)
#define LDP  (BN + 8)

// LDS byte offsets
#define OFF_Q     0
#define OFF_KV    (BM * LDQ * 2)                       // K tile aliases V^T
#define KV_BYTES  (DIM * LDV * 2)                      // V^T is the bigger user
#define OFF_P     (OFF_KV + KV_BYTES)
#define OFF_RMAX  (OFF_P + BM * LDP * 2)
#define OFF_RSUM  (OFF_RMAX + BM * 4 * 4)
#define OFF_M     (OFF_RSUM + BM * 4 * 4)
#define OFF_L     (OFF_M + BM * 4)
#define OFF_RS    (OFF_L + BM * 4)
#define SMEM_BYTES (OFF_RS + BM * 4)                   // 291584

union AB { v16h v; v8h h[2]; };

#if HAVE_TDM
// One 2D tensor->LDS DMA: tile_d1 rows of tile_d0 elements (2 bytes each),
// row pitch stride0 elements, LDS padding per (pad_interval) data.
__device__ __forceinline__ void tdm_load_2d(unsigned lds_addr, const void* gaddr,
                                            unsigned tensor_d0, unsigned tensor_d1,
                                            unsigned tile_d0, unsigned tile_d1,
                                            unsigned long long stride0,
                                            unsigned pad_interval, unsigned pad_amount) {
  unsigned long long ga = (unsigned long long)(uintptr_t)gaddr;
  v4u g0;
  g0.x = 0x1u;                                     // count=1, user descriptor
  g0.y = lds_addr;                                 // LDS byte address
  g0.z = (unsigned)(ga & 0xffffffffu);             // global_addr[31:0]
  g0.w = (unsigned)((ga >> 32) & 0x01ffffffu) | (2u << 30);  // addr[56:32]|type=2
  v8i g1;
  g1[0] = (int)((1u << 16)                         // data_size = 2 bytes
              | (1u << 20)                         // pad_enable
              | (pad_interval << 22)
              | (pad_amount << 25));
  g1[1] = (int)((tensor_d0 & 0xffffu) << 16);      // atomic_addr=0 | td0[15:0]
  g1[2] = (int)(((tensor_d0 >> 16) & 0xffffu) | ((tensor_d1 & 0xffffu) << 16));
  g1[3] = (int)(((tensor_d1 >> 16) & 0xffffu) | ((tile_d0 & 0xffffu) << 16));
  g1[4] = (int)(tile_d1 & 0xffffu);                // tile_dim1 | tile_dim2=0
  g1[5] = (int)(unsigned)(stride0 & 0xffffffffu);  // tensor_dim0_stride[31:0]
  g1[6] = (int)(unsigned)((stride0 >> 32) & 0xffffu);
  g1[7] = 0;                                       // tensor_dim1_stride = 0 (2D)
  v4i gz = {0, 0, 0, 0};
#if __clang_major__ >= 23
  v8i gz8 = {0, 0, 0, 0, 0, 0, 0, 0};
  __builtin_amdgcn_tensor_load_to_lds(g0, g1, gz, gz, gz8, 0);
#else
  __builtin_amdgcn_tensor_load_to_lds(g0, g1, gz, gz, 0);
#endif
}
#endif

// ---------------------------------------------------------------- fp32 -> f16
__global__ __launch_bounds__(256) void cvt_f32_to_f16(
    const float* __restrict__ q, const float* __restrict__ k,
    half_t* __restrict__ qh, half_t* __restrict__ kh) {
  size_t i = ((size_t)blockIdx.x * blockDim.x + threadIdx.x) * 4;
  v4f a = *(const v4f*)(q + i);
  v4f b = *(const v4f*)(k + i);
  v4h ha, hb;
#pragma unroll
  for (int e = 0; e < 4; ++e) { ha[e] = (half_t)a[e]; hb[e] = (half_t)b[e]; }
  *(v4h*)(qh + i) = ha;
  *(v4h*)(kh + i) = hb;
}

// -------------------------------------- V: fp32 [SEQ][DIM] -> f16 [DIM][SEQ]
__global__ __launch_bounds__(256) void transpose_v_f16(
    const float* __restrict__ v, half_t* __restrict__ vt) {
  __shared__ half_t tile[64][64 + 8];
  const int sb = blockIdx.x;            // seq block
  const int db = blockIdx.y;            // dim block
  const int t = threadIdx.x;
  {
    const int r = t >> 4, c4 = (t & 15) * 4;
#pragma unroll
    for (int j = 0; j < 4; ++j) {
      const int row = r + j * 16;
      v4f x = *(const v4f*)(v + (size_t)(sb * 64 + row) * DIM + db * 64 + c4);
#pragma unroll
      for (int e = 0; e < 4; ++e) tile[c4 + e][row] = (half_t)x[e];
    }
  }
  __syncthreads();
  {
    const int rr = t >> 3, cb = (t & 7) * 8;
#pragma unroll
    for (int j = 0; j < 2; ++j) {
      const int dr = rr + j * 32;
      v8h y;
#pragma unroll
      for (int e = 0; e < 8; ++e) y[e] = tile[dr][cb + e];
      *(v8h*)(vt + (size_t)(db * 64 + dr) * SEQ + sb * 64 + cb) = y;
    }
  }
}

// ------------------------------------------------- fused flash attention (partial)
__global__ __launch_bounds__(NTHR, 1) void attn_partial(
    const half_t* __restrict__ Qh, const half_t* __restrict__ Kh,
    const half_t* __restrict__ VhT, float* __restrict__ Opart,
    float* __restrict__ Mstat, float* __restrict__ Lstat) {
  extern __shared__ char smem[];
  half_t* Qs = (half_t*)(smem + OFF_Q);
  half_t* Ks = (half_t*)(smem + OFF_KV);   // K tile, row-major (key, dim)
  half_t* Vt = (half_t*)(smem + OFF_KV);   // V tile transposed (dim, key)
  half_t* Ps = (half_t*)(smem + OFF_P);    // probabilities (row, key) f16
  float* rowmax_p = (float*)(smem + OFF_RMAX);
  float* rowsum_p = (float*)(smem + OFF_RSUM);
  float* m_run    = (float*)(smem + OFF_M);
  float* l_run    = (float*)(smem + OFF_L);
  float* rscale   = (float*)(smem + OFF_RS);

  const int tid  = threadIdx.x;
  const int lane = tid & 31;
  const int wave = tid >> 5;
  const int lo   = lane & 15;
  const int hi   = lane >> 4;
  const int qb   = blockIdx.x;     // query block
  const int sp   = blockIdx.y;     // kv split
  const int r  = wave >> 2, c = wave & 3;   // S-phase tile
  const int r2 = wave & 3, dt = wave >> 2;  // PV-phase slab

  // ---- stage Q block (64 x 1024 f16) into LDS
#if HAVE_TDM
  if (wave == 0) {
    tdm_load_2d((unsigned)(uintptr_t)(void*)Qs,
                Qh + (size_t)qb * BM * DIM,
                /*tensor_d0=*/DIM, /*tensor_d1=*/SEQ,
                /*tile_d0=*/DIM, /*tile_d1=*/BM,
                /*stride0=*/DIM, /*pad_interval=*/7, /*pad_amount=*/1);
  }
#else
  {
    const int row = tid >> 3, colb = (tid & 7) * 8;
    const half_t* gq = Qh + (size_t)(qb * BM + row) * DIM;
    half_t* dq = Qs + row * LDQ;
#pragma unroll 4
    for (int j = 0; j < 16; ++j) {
      const int col = colb + j * 64;
      const int adj = (col >= 512) ? 4 : 0;
      *(v8h*)(dq + col + adj) = *(const v8h*)(gq + col);
    }
  }
#endif
  if (tid < BM) { m_run[tid] = -__builtin_inff(); l_run[tid] = 0.0f; }

  v8f of[16];
#pragma unroll
  for (int i = 0; i < 16; ++i)
#pragma unroll
    for (int e = 0; e < 8; ++e) of[i][e] = 0.0f;

  const size_t keybase0 = (size_t)sp * KEYS_PER_SPLIT;

  for (int t = 0; t < NTILES; ++t) {
    const size_t keybase = keybase0 + (size_t)t * BN;
    __syncthreads();  // S1: previous PV done; init visible on t==0

    // ---- stage K tile (64 x 1024 f16)
#if HAVE_TDM
    if (wave == 0) {
      tdm_load_2d((unsigned)(uintptr_t)(void*)Ks,
                  Kh + keybase * DIM,
                  DIM, SEQ, DIM, BN, DIM, 7, 1);
      __builtin_amdgcn_s_wait_tensorcnt(0);   // covers Q on t==0 as well
    }
#else
    {
      const int row = tid >> 3, colb = (tid & 7) * 8;
      const half_t* gk = Kh + (keybase + row) * DIM;
      half_t* dk = Ks + row * LDK;
#pragma unroll 4
      for (int j = 0; j < 16; ++j) {
        const int col = colb + j * 64;
        const int adj = (col >= 512) ? 4 : 0;
        *(v8h*)(dk + col + adj) = *(const v8h*)(gk + col);
      }
    }
#endif
    __syncthreads();  // S2: K tile (and Q on t==0) ready

    // ---- S = Q_blk @ K_tile^T : one 16x16 tile per wave, K-dim = 1024
    v8f sacc;
#pragma unroll
    for (int e = 0; e < 8; ++e) sacc[e] = 0.0f;
    {
      const half_t* qrow = Qs + (r * 16 + lo) * LDQ;
      const half_t* krow = Ks + (c * 16 + lo) * LDK;
#pragma unroll
      for (int kc = 0; kc < DIM; kc += 32) {
        const int adj = (kc >= 512) ? 4 : 0;   // mid-row LDS pad
        AB a, b;
        a.h[0] = *(const v8h*)(qrow + kc + adj + hi * 8);
        a.h[1] = *(const v8h*)(qrow + kc + adj + 16 + hi * 8);
        b.h[0] = *(const v8h*)(krow + kc + adj + hi * 8);
        b.h[1] = *(const v8h*)(krow + kc + adj + 16 + hi * 8);
        sacc = __builtin_amdgcn_wmma_f32_16x16x32_f16(
            false, a.v, false, b.v, (short)0, sacc, false, false);
      }
    }
    __syncthreads();  // S3: Ks reads done -> region reusable as V^T

    // ---- kick off V^T tile DMA (overlaps the whole softmax phase)
#if HAVE_TDM
    if (wave == 0) {
      tdm_load_2d((unsigned)(uintptr_t)(void*)Vt,
                  VhT + keybase,
                  /*tensor_d0=*/SEQ, /*tensor_d1=*/DIM,
                  /*tile_d0=*/BN, /*tile_d1=*/DIM,
                  /*stride0=*/SEQ, /*pad_interval=*/4, /*pad_amount=*/3);
    }
#endif

    // ---- per-row tile max (C layout: lane = col n, VGPR v = row v + 8*hi)
#pragma unroll
    for (int v = 0; v < 8; ++v) {
      float m = sacc[v];
      m = fmaxf(m, __shfl_xor(m, 1, 32));
      m = fmaxf(m, __shfl_xor(m, 2, 32));
      m = fmaxf(m, __shfl_xor(m, 4, 32));
      m = fmaxf(m, __shfl_xor(m, 8, 32));
      if (lo == 0) rowmax_p[(r * 16 + v + 8 * hi) * 4 + c] = m;
    }
    __syncthreads();  // S4: partial row maxima ready

    if (tid < BM) {  // online softmax running-stat update
      const float* rp = rowmax_p + tid * 4;
      float mt = fmaxf(fmaxf(rp[0], rp[1]), fmaxf(rp[2], rp[3]));
      float mo = m_run[tid];
      float mn = fmaxf(mo, mt);
      float sc = __expf(mo - mn);      // 0 on first tile (mo = -inf)
      l_run[tid] *= sc;
      rscale[tid] = sc;
      m_run[tid] = mn;
    }
    __syncthreads();  // S5: m_run / rscale ready

    // ---- P = exp(S - m), store f16 tile + partial row sums
#pragma unroll
    for (int v = 0; v < 8; ++v) {
      const int row = r * 16 + v + 8 * hi;
      float p = __expf(sacc[v] - m_run[row]);
      Ps[row * LDP + c * 16 + lo] = (half_t)p;
      float s = p;
      s += __shfl_xor(s, 1, 32);
      s += __shfl_xor(s, 2, 32);
      s += __shfl_xor(s, 4, 32);
      s += __shfl_xor(s, 8, 32);
      if (lo == 0) rowsum_p[row * 4 + c] = s;
    }

#if !HAVE_TDM
    // ---- fallback: stage V tile transposed into LDS via VGPRs
    {
      const int row = tid >> 3, colb = (tid & 7) * 8;
      const half_t* gv = VhT + keybase + row;  // unused layout in fallback
      (void)gv;
      // VhT is [DIM][SEQ]: read 8 consecutive seq entries per dim row
      const int dr = tid >> 3;                 // reuse threads: 64 dim rows/pass
#pragma unroll 2
      for (int j = 0; j < 16; ++j) {
        const int dimrow = dr + j * 64;
        const v8h x = *(const v8h*)(VhT + (size_t)dimrow * SEQ + keybase + colb);
        *(v8h*)(Vt + dimrow * LDV + colb) = x;
      }
    }
#endif

    // ---- rescale O accumulators by exp(m_old - m_new)
#pragma unroll
    for (int v = 0; v < 8; ++v) {
      const float sc = rscale[r2 * 16 + v + 8 * hi];
#pragma unroll
      for (int ct = 0; ct < 16; ++ct) of[ct][v] *= sc;
    }

#if HAVE_TDM
    if (wave == 0) __builtin_amdgcn_s_wait_tensorcnt(0);  // V^T landed
#endif
    __syncthreads();  // S6: Ps, Vt, rowsum ready

    if (tid < BM) {
      const float* rp = rowsum_p + tid * 4;
      l_run[tid] += rp[0] + rp[1] + rp[2] + rp[3];
    }

    // ---- O += P @ V_tile  (per wave: 16 rows x 256 dims)
    {
      const half_t* prow = Ps + (r2 * 16 + lo) * LDP;
#pragma unroll
      for (int kch = 0; kch < BN; kch += 32) {
        AB a;
        a.h[0] = *(const v8h*)(prow + kch + hi * 16);
        a.h[1] = *(const v8h*)(prow + kch + hi * 16 + 8);
#pragma unroll
        for (int ct = 0; ct < 16; ++ct) {
          const half_t* vrow = Vt + (dt * 256 + ct * 16 + lo) * LDV + kch + hi * 16;
          AB b;
          b.h[0] = *(const v8h*)(vrow);
          b.h[1] = *(const v8h*)(vrow + 8);
          of[ct] = __builtin_amdgcn_wmma_f32_16x16x32_f16(
              false, a.v, false, b.v, (short)0, of[ct], false, false);
        }
      }
    }
  }

  // ---- epilogue: unnormalized partial O + (m, l) stats for this split
  float* obase = Opart + ((size_t)sp * SEQ + (size_t)qb * BM) * DIM;
#pragma unroll
  for (int v = 0; v < 8; ++v) {
    const int row = r2 * 16 + v + 8 * hi;
#pragma unroll
    for (int ct = 0; ct < 16; ++ct) {
      const int d = dt * 256 + ct * 16 + lo;
      obase[(size_t)row * DIM + d] = of[ct][v];
    }
  }
  if (tid < BM) {
    const size_t si = (size_t)sp * SEQ + (size_t)qb * BM + tid;
    Mstat[si] = m_run[tid];
    Lstat[si] = l_run[tid];
  }
}

// ---------------------------------------------------------------- split merge
__global__ __launch_bounds__(256) void attn_combine(
    const float* __restrict__ Opart, const float* __restrict__ Mstat,
    const float* __restrict__ Lstat, float* __restrict__ out) {
  const int row = blockIdx.x;
  float M = -__builtin_inff();
#pragma unroll
  for (int s = 0; s < KSPLIT; ++s) M = fmaxf(M, Mstat[(size_t)s * SEQ + row]);
  float w[KSPLIT];
  float denom = 0.0f;
#pragma unroll
  for (int s = 0; s < KSPLIT; ++s) {
    w[s] = __expf(Mstat[(size_t)s * SEQ + row] - M);
    denom += w[s] * Lstat[(size_t)s * SEQ + row];
  }
  const float inv = 1.0f / denom;
  for (int d = threadIdx.x; d < DIM; d += 256) {
    float acc = 0.0f;
#pragma unroll
    for (int s = 0; s < KSPLIT; ++s)
      acc += w[s] * Opart[((size_t)s * SEQ + row) * DIM + d];
    out[(size_t)row * DIM + d] = acc * inv;
  }
}

// ---------------------------------------------------------------- launcher
extern "C" void kernel_launch(void* const* d_in, const int* in_sizes, int n_in,
                              void* d_out, int out_size, void* d_ws, size_t ws_size,
                              hipStream_t stream) {
  (void)in_sizes; (void)n_in; (void)out_size; (void)ws_size;
  const float* Q = (const float*)d_in[0];
  const float* K = (const float*)d_in[1];
  const float* V = (const float*)d_in[2];
  float* out = (float*)d_out;

  char* ws = (char*)d_ws;
  const size_t HB = (size_t)SEQ * DIM * sizeof(half_t);   // 8 MB
  half_t* Qh  = (half_t*)(ws);
  half_t* Kh  = (half_t*)(ws + HB);
  half_t* VhT = (half_t*)(ws + 2 * HB);                   // [DIM][SEQ] f16
  float* Opart = (float*)(ws + 3 * HB);                   // KSPLIT * 16 MB
  float* Mstat = (float*)(ws + 3 * HB + (size_t)KSPLIT * SEQ * DIM * sizeof(float));
  float* Lstat = Mstat + (size_t)KSPLIT * SEQ;

  hipFuncSetAttribute((const void*)attn_partial,
                      hipFuncAttributeMaxDynamicSharedMemorySize, SMEM_BYTES);

  cvt_f32_to_f16<<<(SEQ * DIM) / (256 * 4), 256, 0, stream>>>(Q, K, Qh, Kh);
  transpose_v_f16<<<dim3(SEQ / 64, DIM / 64), 256, 0, stream>>>(V, VhT);
  attn_partial<<<dim3(SEQ / BM, KSPLIT), NTHR, SMEM_BYTES, stream>>>(
      Qh, Kh, VhT, Opart, Mstat, Lstat);
  attn_combine<<<SEQ, 256, 0, stream>>>(Opart, Mstat, Lstat, out);
}
